// Metacontroller_18846316495231
// MI455X (gfx1250) — matrix operand order repaired and checked by reading, not compiled, not run on hardware
//
#include <hip/hip_runtime.h>

// ---------------------------------------------------------------------------
// Metacontroller forward for MI455X (gfx1250, wave32, WMMA + TDM).
// All heavy matmuls on v_wmma_f32_16x16x32_bf16 with pre-converted bf16
// operands (convert once, stream b128); scans fused in LDS (24 waves x 1 tile,
// spill-free); activations on single-TRANS hardware approx ops; decoder
// hypernet fully fused; dec2_w A-half staged to LDS via tensor_load_to_lds.
// ---------------------------------------------------------------------------

#define B_DIM 16
#define T_DIM 512
#define D_DIM 256
#define HHD   128      // per-direction hidden
#define G_DIM 32
#define Z_DIM 8
#define R_DIM 16
#define NA_DIM 18
#define DRR   4096     // D_DIM * R_DIM

typedef __attribute__((ext_vector_type(16))) __bf16 v16bf;
typedef __attribute__((ext_vector_type(8)))  __bf16 v8bf;
typedef __attribute__((ext_vector_type(8)))  float  v8f;
typedef __attribute__((ext_vector_type(4)))  unsigned u32x4;
typedef __attribute__((ext_vector_type(8)))  int      i32x8;
typedef __attribute__((ext_vector_type(4)))  int      i32x4;

__device__ __forceinline__ __bf16 f2bf(float f) { return (__bf16)f; }
__device__ __forceinline__ float  bf2f(__bf16 b) { return (float)b; }
// single-TRANS-op activations (v_rcp_f32 / v_exp_f32 / v_rsq_f32, co-exec w/ WMMA)
__device__ __forceinline__ float frcp(float x) { return __builtin_amdgcn_rcpf(x); }
__device__ __forceinline__ float sigm(float x) { return frcp(1.f + __expf(-x)); }
__device__ __forceinline__ float ftanh(float x) {
  float t = __expf(-2.f * x);
  return (1.f - t) * frcp(1.f + t);
}

// A-matrix (16xK 16-bit) lane K pattern (ISA 7.12.2): lanes0-15 K={0..7,16..23},
// lanes16-31 K={8..15,24..31}  -> element i at K = i + half*8 + (i&8)
__device__ __forceinline__ int kA(int i, int half) { return i + half * 8 + (i & 8); }

__device__ __forceinline__ v16bf ldfragA(const __bf16* row, int k0, int half) {
  v8bf lo = *(const v8bf*)(row + k0 + half * 8);        // K = k0+half*8   .. +7
  v8bf hi = *(const v8bf*)(row + k0 + 16 + half * 8);   // K = k0+16+half*8.. +7
  return __builtin_shufflevector(lo, hi, 0, 1, 2, 3, 4, 5, 6, 7,
                                 8, 9, 10, 11, 12, 13, 14, 15);
}
// B-matrix (Kx16 16-bit): lanes0-15 K={0..7,16..23}, lanes16-31 K={8..15,24..31} (row-major W(N,K))
__device__ __forceinline__ v16bf ldfragB(const __bf16* row, int k0, int half) {
  v8bf lo = *(const v8bf*)(row + k0 + half * 8);
  v8bf hi = *(const v8bf*)(row + k0 + 16 + half * 8);
  return __builtin_shufflevector(lo, hi, 0, 1, 2, 3, 4, 5, 6, 7,
                                 8, 9, 10, 11, 12, 13, 14, 15);
}

// ---------------------------------------------------------------------------
// Conversion pre-passes (run once, off the critical loops)
// ---------------------------------------------------------------------------
__global__ void cvt_slice_k(__bf16* __restrict__ dst, const float* __restrict__ src,
                            int rows, int cols, int srcld) {
  int i = blockIdx.x * 256 + threadIdx.x;
  if (i >= rows * cols) return;
  int r = i / cols, c = i - r * cols;
  dst[i] = f2bf(src[(size_t)r * srcld + c]);
}
__global__ void cvt_pad_k(__bf16* __restrict__ dst, const float* __restrict__ src,
                          int rows, int scols, int dcols) {
  int i = blockIdx.x * 256 + threadIdx.x;
  if (i >= rows * dcols) return;
  int r = i / dcols, c = i - r * dcols;
  dst[i] = (c < scols) ? f2bf(src[(size_t)r * scols + c]) : (__bf16)0.f;
}
// e_seq (B,T,D) f32 -> (T*B, D) bf16  (rows m = t*B + b)
__global__ void cvt_e_tb_k(__bf16* __restrict__ dst, const float* __restrict__ src) {
  int i = blockIdx.x * 256 + threadIdx.x;           // over T*B*D
  int k = i & (D_DIM - 1);
  int m = i >> 8;
  int t = m >> 4, b = m & 15;
  dst[i] = f2bf(src[((size_t)b * T_DIM + t) * D_DIM + k]);
}
// dec2_b B-half (d,r) -> transposed bf16 table (r, d) for WMMA B operand
__global__ void cvt_b2bT_k(__bf16* __restrict__ dst, const float* __restrict__ dec2_b) {
  int i = blockIdx.x * 256 + threadIdx.x;           // over R*D
  if (i >= R_DIM * D_DIM) return;
  int r = i >> 8, d = i & (D_DIM - 1);
  dst[i] = f2bf(dec2_b[DRR + d * 16 + r]);
}

// ---------------------------------------------------------------------------
// Specialized bf16-WMMA GEMM: C(M,N) = act( [A1|A2](M,K) @ W(N,K)^T + bias )
// K = KCH*32; K-chunks [0,KSPLIT) from A1, rest from A2.
// Block = 256 threads: A tile (16xK) staged once in LDS, 8 waves x 1 N-tile.
// ---------------------------------------------------------------------------
template <int KCH, int KSPLIT, bool TANH, bool OUTBF>
__global__ void wmma_gemm_k(const __bf16* __restrict__ A1, int lda1,
                            const __bf16* __restrict__ A2, int lda2,
                            const __bf16* __restrict__ W, const float* __restrict__ bias,
                            float* __restrict__ Cf, __bf16* __restrict__ Cb, int N)
{
  constexpr int K = KCH * 32;
  __shared__ __align__(16) __bf16 a_lds[16 * K];
  int tm = blockIdx.y;
  int tid = threadIdx.x;
  int wave = tid >> 5, lane = tid & 31;
  int half = lane >> 4, idx = lane & 15;
  // cooperative A-tile stage: 16 rows x K cols as v8bf copies
  for (int v = tid; v < 16 * (K / 8); v += 256) {
    int r = v / (K / 8), c = (v - r * (K / 8)) * 8;
    const __bf16* src = (c < KSPLIT * 32)
        ? A1 + (size_t)(tm * 16 + r) * lda1 + c
        : A2 + (size_t)(tm * 16 + r) * lda2 + (c - KSPLIT * 32);
    *(v8bf*)(a_lds + r * K + c) = *(const v8bf*)src;
  }
  __syncthreads();
  int tn = blockIdx.x * 8 + wave;
  if (tn * 16 >= N) return;                           // wave-uniform
  const __bf16* wrow = W + (size_t)(tn * 16 + idx) * K;
  v8f acc = {};
#pragma unroll
  for (int kc = 0; kc < KCH; ++kc) {
    v16bf af = ldfragA(a_lds + idx * K, kc * 32, half);
    v16bf bf = ldfragB(wrow, kc * 32, half);
    acc = __builtin_amdgcn_wmma_f32_16x16x32_bf16(false, af, false, bf, (short)0, acc,
                                                  false, false);
  }
  int col = tn * 16 + idx;
  float bv = bias ? bias[col] : 0.f;
#pragma unroll
  for (int j = 0; j < 8; ++j) {
    int row = tm * 16 + j + half * 8;
    float v = acc[j] + bv;
    if (TANH) v = ftanh(v);
    if constexpr (OUTBF) Cb[(size_t)row * N + col] = f2bf(v);
    else                 Cf[(size_t)row * N + col] = v;
  }
}

// ---------------------------------------------------------------------------
// Bidirectional GRU scan. grid=(2) dir, block=768 (24 waves, 1 N-tile each:
// bw[4]+afr[4]+acc ~ 80 VGPRs -> no spills).
// h kept in LDS as f32 (gate math) + bf16 mirror (WMMA fragments via b128 ds).
// ---------------------------------------------------------------------------
__global__ void bigru_scan_k(const float* __restrict__ gi_f, const float* __restrict__ gi_b,
                             const __bf16* __restrict__ whhf_bf, const __bf16* __restrict__ whhb_bf,
                             const float* __restrict__ bhh_f, const float* __restrict__ bhh_b,
                             __bf16* __restrict__ s_bf)
{
  const int dir = blockIdx.x;
  const float*  gi  = dir ? gi_b    : gi_f;
  const __bf16* whh = dir ? whhb_bf : whhf_bf;
  const float*  bhh = dir ? bhh_b   : bhh_f;
  __shared__ float hf_lds[B_DIM * HHD];
  __shared__ __align__(16) __bf16 hb_lds[B_DIM * HHD];
  __shared__ float gh_lds[B_DIM * 3 * HHD];
  int tid = threadIdx.x;
  int wave = tid >> 5, lane = tid & 31;          // 24 waves, wave == N-tile
  int half = lane >> 4, idx = lane & 15;
  for (int i = tid; i < B_DIM * HHD; i += 768) { hf_lds[i] = 0.f; hb_lds[i] = (__bf16)0.f; }
  v16bf bw[4];
#pragma unroll
  for (int kc = 0; kc < 4; ++kc)
    bw[kc] = ldfragB(whh + (size_t)(wave * 16 + idx) * HHD, kc * 32, half);
  __syncthreads();
  for (int step = 0; step < T_DIM; ++step) {
    int t = dir ? (T_DIM - 1 - step) : step;
    v16bf afr[4];
#pragma unroll
    for (int kc = 0; kc < 4; ++kc)
      afr[kc] = ldfragA(hb_lds + idx * HHD, kc * 32, half);
    v8f acc = {};
#pragma unroll
    for (int kc = 0; kc < 4; ++kc)
      acc = __builtin_amdgcn_wmma_f32_16x16x32_bf16(false, afr[kc], false, bw[kc],
                                                    (short)0, acc, false, false);
#pragma unroll
    for (int j = 0; j < 8; ++j)
      gh_lds[(j + half * 8) * (3 * HHD) + wave * 16 + idx] = acc[j];
    __syncthreads();
    for (int e = tid; e < B_DIM * HHD; e += 768) {
      int b = e >> 7, hh = e & (HHD - 1);
      const float* gi_t = gi + ((size_t)t * B_DIM + b) * (3 * HHD);
      float ghr = gh_lds[b * (3 * HHD) + hh]           + bhh[hh];
      float ghz = gh_lds[b * (3 * HHD) + HHD + hh]     + bhh[HHD + hh];
      float ghn = gh_lds[b * (3 * HHD) + 2 * HHD + hh] + bhh[2 * HHD + hh];
      float r = sigm(gi_t[hh] + ghr);
      float z = sigm(gi_t[HHD + hh] + ghz);
      float n = ftanh(gi_t[2 * HHD + hh] + r * ghn);
      float hnew = (1.f - z) * n + z * hf_lds[e];
      hf_lds[e] = hnew;
      hb_lds[e] = f2bf(hnew);
      s_bf[((size_t)t * B_DIM + b) * D_DIM + dir * HHD + hh] = f2bf(hnew);
    }
    __syncthreads();
  }
}

// ---------------------------------------------------------------------------
// Controller scan: encoder GRUCell + heads + switching unit. 1 block, 8 waves.
// ---------------------------------------------------------------------------
__global__ void controller_scan_k(const float* __restrict__ gi_e, const float* __restrict__ swpre,
                                  const float* __restrict__ eps,
                                  const __bf16* __restrict__ whhe_bf, const float* __restrict__ bhh_e,
                                  const float* __restrict__ mu_w, const float* __restrict__ mu_b,
                                  const float* __restrict__ lv_w, const float* __restrict__ lv_b,
                                  const float* __restrict__ sw1_w, const float* __restrict__ sw1_b,
                                  const float* __restrict__ sw2_w, const float* __restrict__ sw2_b,
                                  float* __restrict__ mu_out, float* __restrict__ lv_out,
                                  float* __restrict__ beta_out, float* __restrict__ z_out)
{
  __shared__ float hf_lds[B_DIM * G_DIM];
  __shared__ __align__(16) __bf16 hb_lds[B_DIM * G_DIM];
  __shared__ float gh_lds[B_DIM * 3 * G_DIM];
  __shared__ float zp_lds[B_DIM * Z_DIM];
  __shared__ float mu_lds[B_DIM * Z_DIM];
  __shared__ float lv_lds[B_DIM * Z_DIM];
  __shared__ float hid_lds[B_DIM * 64];
  __shared__ float beta_lds[B_DIM];
  int tid = threadIdx.x;
  int wave = tid >> 5, lane = tid & 31;
  int half = lane >> 4, idx = lane & 15;
  for (int i = tid; i < B_DIM * G_DIM; i += 256) { hf_lds[i] = 0.f; hb_lds[i] = (__bf16)0.f; }
  for (int i = tid; i < B_DIM * Z_DIM; i += 256) zp_lds[i] = 0.f;
  int tnw = (wave < 6) ? wave : 0;
  v16bf bw = ldfragB(whhe_bf + (size_t)(tnw * 16 + idx) * G_DIM, 0, half);
  __syncthreads();
  for (int t = 0; t < T_DIM; ++t) {
    if (wave < 6) {
      v16bf af = ldfragA(hb_lds + idx * G_DIM, 0, half);
      v8f acc = {};
      acc = __builtin_amdgcn_wmma_f32_16x16x32_bf16(false, af, false, bw, (short)0, acc,
                                                    false, false);
#pragma unroll
      for (int j = 0; j < 8; ++j)
        gh_lds[(j + half * 8) * (3 * G_DIM) + wave * 16 + idx] = acc[j];
    }
    __syncthreads();
    for (int e = tid; e < B_DIM * G_DIM; e += 256) {
      int b = e >> 5, g = e & 31;
      const float* gi_t = gi_e + ((size_t)t * B_DIM + b) * (3 * G_DIM);
      float ghr = gh_lds[b * 96 + g]      + bhh_e[g];
      float ghz = gh_lds[b * 96 + 32 + g] + bhh_e[32 + g];
      float ghn = gh_lds[b * 96 + 64 + g] + bhh_e[64 + g];
      float r = sigm(gi_t[g] + ghr);
      float z = sigm(gi_t[32 + g] + ghz);
      float n = ftanh(gi_t[64 + g] + r * ghn);
      float hnew = (1.f - z) * n + z * hf_lds[e];
      hf_lds[e] = hnew;
      hb_lds[e] = f2bf(hnew);
    }
    __syncthreads();
    if (tid < 128) {                       // mu / logvar heads (16 x 8)
      int b = tid >> 3, j = tid & 7;
      float mv = mu_b[j], l = lv_b[j];
      for (int g = 0; g < G_DIM; ++g) {
        float hv = hf_lds[b * G_DIM + g];
        mv += hv * mu_w[j * G_DIM + g];
        l  += hv * lv_w[j * G_DIM + g];
      }
      mu_lds[b * 8 + j] = mv; lv_lds[b * 8 + j] = l;
      size_t o = ((size_t)b * T_DIM + t) * Z_DIM + j;
      mu_out[o] = mv; lv_out[o] = l;
    }
    for (int e = tid; e < B_DIM * 64; e += 256) {   // switching hidden
      int b = e >> 6, o = e & 63;
      float a = sw1_b[o] + swpre[((size_t)t * B_DIM + b) * 64 + o];
      const float* wrow = sw1_w + (size_t)o * 296;
      for (int g = 0; g < G_DIM; ++g)   a += hf_lds[b * G_DIM + g] * wrow[256 + g];
      for (int zi = 0; zi < Z_DIM; ++zi) a += zp_lds[b * 8 + zi] * wrow[288 + zi];
      hid_lds[e] = ftanh(a);
    }
    __syncthreads();
    if (tid < B_DIM) {
      float a = sw2_b[0];
      for (int o = 0; o < 64; ++o) a += hid_lds[tid * 64 + o] * sw2_w[o];
      float be = sigm(a);
      beta_lds[tid] = be;
      beta_out[(size_t)tid * T_DIM + t] = be;
    }
    __syncthreads();
    if (tid < 128) {
      int b = tid >> 3, j = tid & 7;
      float be = beta_lds[b];
      float zv = be * (mu_lds[b * 8 + j] +
                       __expf(0.5f * lv_lds[b * 8 + j]) * eps[((size_t)b * T_DIM + t) * Z_DIM + j])
               + (1.f - be) * zp_lds[b * 8 + j];
      z_out[((size_t)b * T_DIM + t) * Z_DIM + j] = zv;
      zp_lds[b * 8 + j] = zv;
    }
    __syncthreads();
  }
}

// ---------------------------------------------------------------------------
// Decoder stage 1: tmp[m,r] = sum_d Bm[m,d,r]*e[m,d].
// Main term: 256 chained accumulate-in-C WMMAs (A = hmid scaled by e[row,d]).
// Bias term: 8 chained WMMAs of E(16x256) @ biasB^T(16x256).
// ---------------------------------------------------------------------------
__global__ void dec_tmp_k(const __bf16* __restrict__ hmid_bf, const __bf16* __restrict__ e_tb,
                          const __bf16* __restrict__ dec2w_bf, const __bf16* __restrict__ b2bT,
                          float* __restrict__ tmp)
{
  int m0 = blockIdx.x * 16;
  int lane = threadIdx.x, half = lane >> 4, idx = lane & 15;
  int m = m0 + idx;                                   // output row (b*T + t)
  const __bf16* erow = e_tb + (size_t)((m & (T_DIM - 1)) * B_DIM + (m >> 9)) * D_DIM;
  float hbase[16];
#pragma unroll
  for (int i = 0; i < 16; ++i)
    hbase[i] = bf2f(hmid_bf[(size_t)m * G_DIM + kA(i, half)]);
  v8f acc = {};
  for (int d = 0; d < D_DIM; ++d) {
    if (d + 1 < D_DIM)
      __builtin_prefetch(dec2w_bf + (size_t)(DRR + (d + 1) * 16 + idx) * G_DIM, 0, 1);
    float ev = bf2f(erow[d]);
    v16bf af;
#pragma unroll
    for (int i = 0; i < 16; ++i) af[i] = f2bf(hbase[i] * ev);
    v16bf bfr = ldfragB(dec2w_bf + (size_t)(DRR + d * 16 + idx) * G_DIM, 0, half);
    acc = __builtin_amdgcn_wmma_f32_16x16x32_bf16(false, af, false, bfr, (short)0, acc,
                                                  false, false);
  }
  v8f accb = {};
#pragma unroll
  for (int kc = 0; kc < 8; ++kc) {
    v16bf af = ldfragA(erow, kc * 32, half);
    v16bf bfr = ldfragB(b2bT + (size_t)idx * D_DIM, kc * 32, half);
    accb = __builtin_amdgcn_wmma_f32_16x16x32_bf16(false, af, false, bfr, (short)0, accb,
                                                   false, false);
  }
#pragma unroll
  for (int j = 0; j < 8; ++j) {
    int row = m0 + j + half * 8;
    tmp[(size_t)row * R_DIM + idx] = acc[j] + accb[j];
  }
}

// ---------------------------------------------------------------------------
// Decoder stage 2: perturbed = e + A @ tmp, then LayerNorm + action head.
// dec2_w A-half staged into LDS with the Tensor Data Mover (32 KB chunks),
// consumed by WMMA from LDS; cross-lane reduce over r via shfl_xor.
// ---------------------------------------------------------------------------
__global__ void dec_apply_k(const __bf16* __restrict__ hmid_bf, const float* __restrict__ e_seq,
                            const __bf16* __restrict__ dec2w_bf, const float* __restrict__ dec2_b,
                            const float* __restrict__ tmp,
                            const float* __restrict__ ln_g, const float* __restrict__ ln_b,
                            const float* __restrict__ head_w, const float* __restrict__ head_b,
                            float* __restrict__ logits)
{
  __shared__ __align__(16) __bf16 stage[512 * G_DIM];   // 32 KB, first in LDS
  __shared__ float pert[16 * D_DIM];
  __shared__ float mean_s[16], rstd_s[16];
  int m0 = blockIdx.x * 16;
  int tid = threadIdx.x;
  if (tid < 32) {                                       // wave 0: WMMA pass
    int lane = tid, half = lane >> 4, idx = lane & 15;
    v16bf af = ldfragA(hmid_bf + (size_t)(m0 + idx) * G_DIM, 0, half);
    float t8[8];
#pragma unroll
    for (int j = 0; j < 8; ++j)
      t8[j] = tmp[(size_t)(m0 + j + half * 8) * R_DIM + idx];
    for (int c = 0; c < 8; ++c) {
      // --- Tensor Data Mover: stage 512 rows x 32 cols bf16 of dec2_w (A half)
      unsigned long long ga =
          (unsigned long long)(uintptr_t)(dec2w_bf + (size_t)c * 512 * G_DIM);
      u32x4 g0 = (u32x4)0;
      g0[0] = 1u;                                        // count = 1 descriptor
      g0[1] = 0u;                                        // lds_addr = 0 (stage @ base)
      g0[2] = (unsigned)(ga & 0xFFFFFFFFu);
      g0[3] = (unsigned)((ga >> 32) & 0x01FFFFFFu) | (2u << 30);   // type = 2
      i32x8 g1 = (i32x8)0;
      g1[0] = 1 << 16;                                   // data_size = 2 bytes
      g1[1] = 32 << 16;                                  // tensor_dim0 = 32
      g1[2] = 512 << 16;                                 // tensor_dim1 = 512
      g1[3] = 32 << 16;                                  // tile_dim0 = 32
      g1[4] = 512;                                       // tile_dim1 = 512
      g1[5] = 32;                                        // tensor_dim0_stride = 32
      i32x4 gz4 = (i32x4)0;
      i32x8 gz8 = (i32x8)0;
      __builtin_amdgcn_tensor_load_to_lds(g0, g1, gz4, gz4, gz8, 0);
      __builtin_amdgcn_s_wait_tensorcnt(0);
      for (int dl = 0; dl < 32; ++dl) {
        int d = c * 32 + dl;
        v16bf bfr = ldfragB(stage + (size_t)(dl * 16 + idx) * G_DIM, 0, half);
        v8f zero = {};
        v8f acc = __builtin_amdgcn_wmma_f32_16x16x32_bf16(false, af, false, bfr, (short)0,
                                                          zero, false, false);
        float ba = dec2_b[d * 16 + idx];
#pragma unroll
        for (int j = 0; j < 8; ++j) {
          float v = (acc[j] + ba) * t8[j];               // A[row,d,r] * tmp[row,r]
          v += __shfl_xor(v, 1, 32);                     // reduce over r (16-lane halves)
          v += __shfl_xor(v, 2, 32);
          v += __shfl_xor(v, 4, 32);
          v += __shfl_xor(v, 8, 32);
          if (idx == 0) {
            int row = j + half * 8;
            pert[row * D_DIM + d] = e_seq[(size_t)(m0 + row) * D_DIM + d] + v;
          }
        }
      }
    }
  }
  __syncthreads();
  if (tid < 16) {                                        // LayerNorm statistics
    float mn = 0.f;
    for (int d = 0; d < D_DIM; ++d) mn += pert[tid * D_DIM + d];
    mn *= (1.f / D_DIM);
    float vr = 0.f;
    for (int d = 0; d < D_DIM; ++d) { float df = pert[tid * D_DIM + d] - mn; vr += df * df; }
    vr *= (1.f / D_DIM);
    mean_s[tid] = mn;
    rstd_s[tid] = __builtin_amdgcn_rsqf(vr + 1e-5f);
  }
  __syncthreads();
  for (int o = tid; o < 16 * NA_DIM; o += blockDim.x) {  // action head
    int row = o / NA_DIM, a = o - row * NA_DIM;
    float mn = mean_s[row], rs = rstd_s[row];
    float accv = head_b[a];
    for (int d = 0; d < D_DIM; ++d) {
      float xn = (pert[row * D_DIM + d] - mn) * rs * ln_g[d] + ln_b[d];
      accv += xn * head_w[(size_t)a * D_DIM + d];
    }
    logits[(size_t)(m0 + row) * NA_DIM + a] = accv;
  }
}

// ---------------------------------------------------------------------------
extern "C" void kernel_launch(void* const* d_in, const int* in_sizes, int n_in,
                              void* d_out, int out_size, void* d_ws, size_t ws_size,
                              hipStream_t stream) {
  (void)in_sizes; (void)n_in; (void)out_size; (void)ws_size;
  const float* e_seq  = (const float*)d_in[0];
  const float* eps    = (const float*)d_in[1];
  const float* wih_f  = (const float*)d_in[2];
  const float* whh_f  = (const float*)d_in[3];
  const float* bih_f  = (const float*)d_in[4];
  const float* bhh_f  = (const float*)d_in[5];
  const float* wih_b  = (const float*)d_in[6];
  const float* whh_b  = (const float*)d_in[7];
  const float* bih_b  = (const float*)d_in[8];
  const float* bhh_b  = (const float*)d_in[9];
  const float* wih_e  = (const float*)d_in[10];
  const float* whh_e  = (const float*)d_in[11];
  const float* bih_e  = (const float*)d_in[12];
  const float* bhh_e  = (const float*)d_in[13];
  const float* mu_w   = (const float*)d_in[14];
  const float* mu_b   = (const float*)d_in[15];
  const float* lv_w   = (const float*)d_in[16];
  const float* lv_b   = (const float*)d_in[17];
  const float* sw1_w  = (const float*)d_in[18];
  const float* sw1_b  = (const float*)d_in[19];
  const float* sw2_w  = (const float*)d_in[20];
  const float* sw2_b  = (const float*)d_in[21];
  const float* dec1_w = (const float*)d_in[22];
  const float* dec1_b = (const float*)d_in[23];
  const float* dec2_w = (const float*)d_in[24];
  const float* dec2_b = (const float*)d_in[25];
  const float* ln_g   = (const float*)d_in[26];
  const float* ln_b   = (const float*)d_in[27];
  const float* head_w = (const float*)d_in[28];
  const float* head_b = (const float*)d_in[29];

  const int M = B_DIM * T_DIM;   // 8192

  // ---- workspace carve-up (256B-aligned regions, ~42 MB total) -------------
  char* p = (char*)d_ws;
  auto alloc = [&](size_t bytes) { char* r = p; p += (bytes + 255) & ~(size_t)255; return r; };
  float*  gi_f   = (float*)alloc((size_t)M * 384 * 4);
  float*  gi_b   = (float*)alloc((size_t)M * 384 * 4);
  float*  gi_e   = (float*)alloc((size_t)M * 96 * 4);
  float*  swpre  = (float*)alloc((size_t)M * 64 * 4);
  float*  tmp    = (float*)alloc((size_t)M * R_DIM * 4);
  __bf16* e_tb   = (__bf16*)alloc((size_t)M * D_DIM * 2);
  __bf16* s_bf   = (__bf16*)alloc((size_t)M * D_DIM * 2);
  __bf16* wihf_bf = (__bf16*)alloc((size_t)384 * 256 * 2);
  __bf16* wihb_bf = (__bf16*)alloc((size_t)384 * 256 * 2);
  __bf16* whhf_bf = (__bf16*)alloc((size_t)384 * 128 * 2);
  __bf16* whhb_bf = (__bf16*)alloc((size_t)384 * 128 * 2);
  __bf16* wihe_bf = (__bf16*)alloc((size_t)96 * 512 * 2);
  __bf16* whhe_bf = (__bf16*)alloc((size_t)96 * 32 * 2);
  __bf16* sw1s_bf = (__bf16*)alloc((size_t)64 * 256 * 2);
  __bf16* dec1p_bf = (__bf16*)alloc((size_t)32 * 32 * 2);
  __bf16* dec2_bf  = (__bf16*)alloc((size_t)2 * DRR * 32 * 2);
  __bf16* b2bT_bf  = (__bf16*)alloc((size_t)R_DIM * D_DIM * 2);
  __bf16* zpad_bf  = (__bf16*)alloc((size_t)M * 32 * 2);
  __bf16* hmid_bf  = (__bf16*)alloc((size_t)M * 32 * 2);

  float* out      = (float*)d_out;
  float* logits   = out;
  float* mu_out   = logits   + (size_t)M * NA_DIM;
  float* lv_out   = mu_out   + (size_t)M * Z_DIM;
  float* beta_out = lv_out   + (size_t)M * Z_DIM;
  float* z_out    = beta_out + (size_t)M;

  auto nb = [](int n) { return (n + 255) / 256; };

  // ---- one-time bf16 conversions ------------------------------------------
  cvt_e_tb_k<<<nb(M * D_DIM), 256, 0, stream>>>(e_tb, e_seq);
  cvt_slice_k<<<nb(384 * 256), 256, 0, stream>>>(wihf_bf, wih_f, 384, 256, 256);
  cvt_slice_k<<<nb(384 * 256), 256, 0, stream>>>(wihb_bf, wih_b, 384, 256, 256);
  cvt_slice_k<<<nb(384 * 128), 256, 0, stream>>>(whhf_bf, whh_f, 384, 128, 128);
  cvt_slice_k<<<nb(384 * 128), 256, 0, stream>>>(whhb_bf, whh_b, 384, 128, 128);
  cvt_slice_k<<<nb(96 * 512), 256, 0, stream>>>(wihe_bf, wih_e, 96, 512, 512);
  cvt_slice_k<<<nb(96 * 32), 256, 0, stream>>>(whhe_bf, whh_e, 96, 32, 32);
  cvt_slice_k<<<nb(64 * 256), 256, 0, stream>>>(sw1s_bf, sw1_w, 64, 256, 296);
  cvt_slice_k<<<nb(2 * DRR * 32), 256, 0, stream>>>(dec2_bf, dec2_w, 2 * DRR, 32, 32);
  cvt_b2bT_k<<<nb(R_DIM * D_DIM), 256, 0, stream>>>(b2bT_bf, dec2_b);
  cvt_pad_k<<<nb(32 * 32), 256, 0, stream>>>(dec1p_bf, dec1_w, 32, 8, 32);

  // ---- time-parallel input-projection GEMMs (256 thr, LDS-staged A) -------
  wmma_gemm_k<8, 8, false, false><<<dim3(3, M / 16), 256, 0, stream>>>(
      e_tb, 256, nullptr, 0, wihf_bf, bih_f, gi_f, nullptr, 384);
  wmma_gemm_k<8, 8, false, false><<<dim3(3, M / 16), 256, 0, stream>>>(
      e_tb, 256, nullptr, 0, wihb_bf, bih_b, gi_b, nullptr, 384);
  wmma_gemm_k<8, 8, false, false><<<dim3(1, M / 16), 256, 0, stream>>>(
      e_tb, 256, nullptr, 0, sw1s_bf, nullptr, swpre, nullptr, 64);
  // ---- bidirectional GRU scan ---------------------------------------------
  bigru_scan_k<<<2, 768, 0, stream>>>(gi_f, gi_b, whhf_bf, whhb_bf, bhh_f, bhh_b, s_bf);
  // ---- controller input projection over concat([e, s]) (split-K) ----------
  wmma_gemm_k<16, 8, false, false><<<dim3(1, M / 16), 256, 0, stream>>>(
      e_tb, 256, s_bf, 256, wihe_bf, bih_e, gi_e, nullptr, 96);
  // ---- controller scan -> mu/lv/beta/z ------------------------------------
  controller_scan_k<<<1, 256, 0, stream>>>(gi_e, swpre, eps, whhe_bf, bhh_e,
                                           mu_w, mu_b, lv_w, lv_b,
                                           sw1_w, sw1_b, sw2_w, sw2_b,
                                           mu_out, lv_out, beta_out, z_out);
  // ---- decoder hypernet mid layer (K padded 8 -> 32) ----------------------
  cvt_pad_k<<<nb(M * 32), 256, 0, stream>>>(zpad_bf, z_out, M, Z_DIM, 32);
  wmma_gemm_k<1, 1, true, true><<<dim3(1, M / 16), 256, 0, stream>>>(
      zpad_bf, 32, nullptr, 0, dec1p_bf, dec1_b, nullptr, hmid_bf, 32);
  // ---- fused decoder (ab tensor never materialized) -----------------------
  dec_tmp_k<<<M / 16, 32, 0, stream>>>(hmid_bf, e_tb, dec2_bf, b2bT_bf, tmp);
  dec_apply_k<<<M / 16, 128, 0, stream>>>(hmid_bf, e_seq, dec2_bf, dec2_b, tmp,
                                          ln_g, ln_b, head_w, head_b, logits);
}
